// MultiHeadAttention_59450937312052
// MI455X (gfx1250) — matrix operand order repaired and checked by reading, not compiled, not run on hardware
//
#include <hip/hip_runtime.h>
#include <hip/hip_bf16.h>
#include <hip/hip_fp16.h>

// MI455X / gfx1250 compile-only target. wave32, WMMA 16x16x32 f16 -> f32.

typedef __attribute__((ext_vector_type(16))) _Float16 v16h;
typedef __attribute__((ext_vector_type(8)))  float    v8f;

#define BATCH  8192
#define DMODEL 1024
#define SCALE  0.125f   // 1/sqrt(HEAD_DIM=64)

// ---------------------------------------------------------------------------
// Fragment loaders (wave32); A and B share one contiguous k order, k = 16*(lane/16) + s.
// ---------------------------------------------------------------------------

// A-matrix 16x32 f16 (MxK), row-major source with leading dim `ld` (elements).
// Lane m = lane&15 holds row m. Lanes 0-15: K 0..15; lanes 16-31: K 16..31.
static __device__ __forceinline__ v16h load_frag_a(const _Float16* p, int ld) {
  int lane = threadIdx.x & 31;
  int m    = lane & 15;
  int kb   = lane & 16;                 // 0 or 16
  const _Float16* row = p + m * ld + kb;
  v16h a;
#pragma unroll
  for (int r = 0; r < 16; ++r) a[r] = row[r];
  return a;
}

// B-matrix 32x16 f16 (KxN) where B[k][n] = src[n*ld + k]  (source rows are the
// N dimension: used for W (x@W^T), K (Q@K^T) and transposed V). Contiguous
// along K -> 2x global_load_b128 per lane.
// Lane n = lane&15 holds column n; lanes 0-15 -> K 0..15, lanes 16-31 -> K 16..31.
static __device__ __forceinline__ v16h load_frag_bt(const _Float16* p, int ld) {
  int lane = threadIdx.x & 31;
  int n    = lane & 15;
  int kb   = lane & 16;                 // 0 or 16
  const _Float16* row = p + n * ld + kb;
  v16h b;
#pragma unroll
  for (int r = 0; r < 16; ++r) b[r] = row[r];
  return b;
}

static __device__ __forceinline__ v8f wmma_f16(v16h a, v16h b, v8f c) {
  // (neg_a, A, neg_b, B, c_mod, C, reuse_a, reuse_b)
  return __builtin_amdgcn_wmma_f32_16x16x32_f16(false, a, false, b, (short)0, c,
                                                false, false);
}

static __device__ __forceinline__ v8f zero8() {
  v8f z = {0.f, 0.f, 0.f, 0.f, 0.f, 0.f, 0.f, 0.f};
  return z;
}

// ---------------------------------------------------------------------------
// f32 -> f16 conversion (grid-stride)
// ---------------------------------------------------------------------------
__global__ void cvt_f32_to_f16(const float* __restrict__ src,
                               _Float16* __restrict__ dst, int n) {
  int stride = gridDim.x * blockDim.x;
  for (int i = blockIdx.x * blockDim.x + threadIdx.x; i < n; i += stride)
    dst[i] = (_Float16)src[i];
}

// ---------------------------------------------------------------------------
// Projections: Out = xh @ W^T + bias, stored f16.  One wave per 16x16 tile.
// Q,K stored row-major [BATCH][DMODEL]; V stored TRANSPOSED [DMODEL][BATCH]
// so that P@V B-fragments are contiguous along K (keys).
// blockIdx: x = col tile (D/16), y = row tile (B/16), z = which of {Q,K,V}.
// ---------------------------------------------------------------------------
__global__ void __launch_bounds__(32)
proj_kernel(const _Float16* __restrict__ xh,
            const _Float16* __restrict__ Wqh, const _Float16* __restrict__ Wkh,
            const _Float16* __restrict__ Wvh,
            const float* __restrict__ bq, const float* __restrict__ bk,
            const float* __restrict__ bv,
            _Float16* __restrict__ Qh, _Float16* __restrict__ Kh,
            _Float16* __restrict__ Vt) {
  int which = blockIdx.z;
  const _Float16* W   = (which == 0) ? Wqh : (which == 1) ? Wkh : Wvh;
  const float*    bia = (which == 0) ? bq  : (which == 1) ? bk  : bv;

  int col0 = blockIdx.x * 16;
  int row0 = blockIdx.y * 16;

  v8f c = zero8();
  for (int kk = 0; kk < DMODEL; kk += 32) {
    v16h a = load_frag_a(xh + row0 * DMODEL + kk, DMODEL);
    v16h b = load_frag_bt(W + col0 * DMODEL + kk, DMODEL);
    c = wmma_f16(a, b, c);
  }

  int lane = threadIdx.x & 31;
  int n    = lane & 15;
  int hi   = (lane >> 4) & 1;
  float bb = bia[col0 + n];

  if (which == 2) {
    // transposed store: Vt[col][row]; 8 consecutive rows -> one b128 store
    _Float16* dst = Vt + (col0 + n) * BATCH + row0 + hi * 8;
#pragma unroll
    for (int r = 0; r < 8; ++r) dst[r] = (_Float16)(c[r] + bb);
  } else {
    _Float16* out = (which == 0) ? Qh : Kh;
#pragma unroll
    for (int r = 0; r < 8; ++r) {
      int row = row0 + hi * 8 + r;
      out[row * DMODEL + col0 + n] = (_Float16)(c[r] + bb);
    }
  }
}

// ---------------------------------------------------------------------------
// Pass A: per-row softmax statistics (max M, denom L) via streaming WMMA.
// One block (8 waves) per 16-query tile; waves split the 8192 keys.
// ---------------------------------------------------------------------------
__global__ void __launch_bounds__(256)
attn_stats_kernel(const _Float16* __restrict__ Qh,
                  const _Float16* __restrict__ Kh,
                  float* __restrict__ Mrow, float* __restrict__ Lrow) {
  __shared__ float sm[8][16];
  __shared__ float sl[8][16];

  int q0   = blockIdx.x * 16;
  int wave = threadIdx.x >> 5;
  int lane = threadIdx.x & 31;
  int hi   = (lane >> 4) & 1;

  float mrun[8], lrun[8];
#pragma unroll
  for (int r = 0; r < 8; ++r) { mrun[r] = -1e30f; lrun[r] = 0.f; }

  for (int c = wave; c < BATCH / 32; c += 8) {     // 32-key chunks
    int k0 = c * 32;
    v8f e0 = zero8();
    v8f e1 = zero8();
    for (int kk = 0; kk < DMODEL; kk += 32) {
      v16h a  = load_frag_a(Qh + q0 * DMODEL + kk, DMODEL);      // shared A
      v16h b0 = load_frag_bt(Kh + k0 * DMODEL + kk, DMODEL);
      v16h b1 = load_frag_bt(Kh + (k0 + 16) * DMODEL + kk, DMODEL);
      e0 = wmma_f16(a, b0, e0);
      e1 = wmma_f16(a, b1, e1);
    }
#pragma unroll
    for (int t = 0; t < 2; ++t) {
      v8f e = t ? e1 : e0;
#pragma unroll
      for (int r = 0; r < 8; ++r) {
        float ev = e[r] * SCALE;
        // row-wise reduce across the 16 lanes of this half (columns)
        float mx = ev;
#pragma unroll
        for (int s = 1; s < 16; s <<= 1) mx = fmaxf(mx, __shfl_xor(mx, s, 32));
        float nm = fmaxf(mrun[r], mx);
        float p  = __expf(ev - nm);
#pragma unroll
        for (int s = 1; s < 16; s <<= 1) p += __shfl_xor(p, s, 32);
        lrun[r] = lrun[r] * __expf(mrun[r] - nm) + p;
        mrun[r] = nm;
      }
    }
  }

  if ((lane & 15) == 0) {
#pragma unroll
    for (int r = 0; r < 8; ++r) {
      sm[wave][hi * 8 + r] = mrun[r];
      sl[wave][hi * 8 + r] = lrun[r];
    }
  }
  __syncthreads();

  if (threadIdx.x < 16) {
    int t = threadIdx.x;
    float M = -1e30f;
    for (int w = 0; w < 8; ++w) M = fmaxf(M, sm[w][t]);
    float L = 0.f;
    for (int w = 0; w < 8; ++w) L += sl[w][t] * __expf(sm[w][t] - M);
    Mrow[q0 + t] = M;
    Lrow[q0 + t] = L;
  }
}

// ---------------------------------------------------------------------------
// Pass B: out = softmax(QK^T * scale) @ V.
// One block (8 waves) per 16-query tile. Per 256-key iteration each wave
// computes the energy tile for its own 32-key chunk, writes exp-weights (f16)
// to LDS; after a barrier every wave multiplies all 8 P chunks against its
// private 128-column slice of V (8 f32 accumulator tiles). V is consumed from
// the transposed copy Vt so B-fragments are contiguous b128 loads.
// ---------------------------------------------------------------------------
__global__ void __launch_bounds__(256)
attn_out_kernel(const _Float16* __restrict__ Qh,
                const _Float16* __restrict__ Kh,
                const _Float16* __restrict__ Vt,
                const float* __restrict__ Mrow, const float* __restrict__ Lrow,
                float* __restrict__ out) {
  __shared__ _Float16 P[8][16][32];   // 8 KB

  int q0   = blockIdx.x * 16;
  int wave = threadIdx.x >> 5;
  int lane = threadIdx.x & 31;
  int n    = lane & 15;
  int hi   = (lane >> 4) & 1;
  int col0 = wave * 128;              // this wave's 128-column V/O slice

  float m8[8];
#pragma unroll
  for (int r = 0; r < 8; ++r) m8[r] = Mrow[q0 + hi * 8 + r];

  v8f acc[8];
#pragma unroll
  for (int ct = 0; ct < 8; ++ct) acc[ct] = zero8();

  for (int it = 0; it < BATCH / 256; ++it) {
    int c  = it * 8 + wave;           // this wave's 32-key chunk
    int k0 = c * 32;
    v8f e0 = zero8();
    v8f e1 = zero8();
    for (int kk = 0; kk < DMODEL; kk += 32) {
      v16h a  = load_frag_a(Qh + q0 * DMODEL + kk, DMODEL);
      v16h b0 = load_frag_bt(Kh + k0 * DMODEL + kk, DMODEL);
      v16h b1 = load_frag_bt(Kh + (k0 + 16) * DMODEL + kk, DMODEL);
      e0 = wmma_f16(a, b0, e0);
      e1 = wmma_f16(a, b1, e1);
    }
#pragma unroll
    for (int t = 0; t < 2; ++t) {
      v8f e = t ? e1 : e0;
#pragma unroll
      for (int r = 0; r < 8; ++r) {
        float p = __expf(e[r] * SCALE - m8[r]);
        P[wave][hi * 8 + r][t * 16 + n] = (_Float16)p;
      }
    }
    __syncthreads();

    for (int ch = 0; ch < 8; ++ch) {
      v16h pa  = load_frag_a(&P[ch][0][0], 32);        // LDS gather (ds_load)
      int key0 = (it * 8 + ch) * 32;
#pragma unroll
      for (int ct = 0; ct < 8; ++ct) {
        // B[k][n] = Vt[(col0+ct*16+n)*BATCH + key0 + k]  (contiguous in k)
        v16h vb = load_frag_bt(Vt + (col0 + ct * 16) * BATCH + key0, BATCH);
        acc[ct] = wmma_f16(pa, vb, acc[ct]);
      }
    }
    __syncthreads();
  }

  float l8[8];
#pragma unroll
  for (int r = 0; r < 8; ++r) l8[r] = Lrow[q0 + hi * 8 + r];

#pragma unroll
  for (int ct = 0; ct < 8; ++ct) {
#pragma unroll
    for (int r = 0; r < 8; ++r) {
      int row = q0 + hi * 8 + r;
      out[row * DMODEL + col0 + ct * 16 + n] = acc[ct][r] / l8[r];
    }
  }
}

// ---------------------------------------------------------------------------
// Launch. Workspace layout (f16 staging, ~73.6 MB total):
//   xh[B*D] | Wqh[D*D] | Wkh | Wvh | Qh[B*D] | Kh[B*D] | Vt[D*B] | Mrow | Lrow
// ---------------------------------------------------------------------------
extern "C" void kernel_launch(void* const* d_in, const int* in_sizes, int n_in,
                              void* d_out, int out_size, void* d_ws,
                              size_t ws_size, hipStream_t stream) {
  const float* x  = (const float*)d_in[0];
  const float* Wq = (const float*)d_in[1];
  const float* bq = (const float*)d_in[2];
  const float* Wk = (const float*)d_in[3];
  const float* bk = (const float*)d_in[4];
  const float* Wv = (const float*)d_in[5];
  const float* bv = (const float*)d_in[6];
  float* out = (float*)d_out;

  char*  ws  = (char*)d_ws;
  size_t off = 0;
  _Float16* xh  = (_Float16*)(ws + off); off += (size_t)BATCH * DMODEL * 2;
  _Float16* Wqh = (_Float16*)(ws + off); off += (size_t)DMODEL * DMODEL * 2;
  _Float16* Wkh = (_Float16*)(ws + off); off += (size_t)DMODEL * DMODEL * 2;
  _Float16* Wvh = (_Float16*)(ws + off); off += (size_t)DMODEL * DMODEL * 2;
  _Float16* Qh  = (_Float16*)(ws + off); off += (size_t)BATCH * DMODEL * 2;
  _Float16* Kh  = (_Float16*)(ws + off); off += (size_t)BATCH * DMODEL * 2;
  _Float16* Vt  = (_Float16*)(ws + off); off += (size_t)BATCH * DMODEL * 2;
  float* Mrow = (float*)(ws + off); off += (size_t)BATCH * 4;
  float* Lrow = (float*)(ws + off); off += (size_t)BATCH * 4;
  (void)ws_size; (void)in_sizes; (void)n_in; (void)out_size;

  cvt_f32_to_f16<<<2048, 256, 0, stream>>>(x, xh, BATCH * DMODEL);
  cvt_f32_to_f16<<<1024, 256, 0, stream>>>(Wq, Wqh, DMODEL * DMODEL);
  cvt_f32_to_f16<<<1024, 256, 0, stream>>>(Wk, Wkh, DMODEL * DMODEL);
  cvt_f32_to_f16<<<1024, 256, 0, stream>>>(Wv, Wvh, DMODEL * DMODEL);

  proj_kernel<<<dim3(DMODEL / 16, BATCH / 16, 3), 32, 0, stream>>>(
      xh, Wqh, Wkh, Wvh, bq, bk, bv, Qh, Kh, Vt);

  attn_stats_kernel<<<BATCH / 16, 256, 0, stream>>>(Qh, Kh, Mrow, Lrow);

  attn_out_kernel<<<BATCH / 16, 256, 0, stream>>>(Qh, Kh, Vt, Mrow, Lrow, out);
}
